// CrossBlock_68350109549135
// MI455X (gfx1250) — matrix operand order repaired and checked by reading, not compiled, not run on hardware
//
#include <hip/hip_runtime.h>
#include <hip/hip_bf16.h>

// ---------------------------------------------------------------------------
// CDNA5 (gfx1250) CrossBlock: bf16 WMMA GEMMs + fp32 softmax / LN / GELU.
// 256-thread blocks = 8 wave32s; block tile 64x64, K-step 64.
// Each wave owns a 16x32 output strip -> 4 v_wmma_f32_16x16x32_bf16 per K-step.
// LDS tiles: A [64][72] (M,K), B transposed [64][72] (N,K); fragments are
// two 16B runs per lane (k 8h.., 16+8h..) -> ds_load_b128 pairs. Weights/V are pre-transposed to
// bf16 [N][K] once per call so B tiles load via global_load_async_to_lds_b128.
// ---------------------------------------------------------------------------

typedef __attribute__((ext_vector_type(16))) __bf16 v16bf;
typedef __attribute__((ext_vector_type(8)))  float  v8f;
typedef __bf16 bf16_t;

#define LDA_P 72   // padded row stride (elems); 144 B = multiple of 16 B
#define LDB_P 72

__device__ __forceinline__ v8f wmma_bf16(v16bf a, v16bf b, v8f c) {
  return __builtin_amdgcn_wmma_f32_16x16x32_bf16(
      /*neg_a=*/false, a, /*neg_b=*/false, b,
      /*c_mod=*/(short)0, c, /*reuse_a=*/false, /*reuse_b=*/false);
}

union frag_u { uint4 q[2]; v16bf v; };
union pk2_u  { bf16_t h[2]; unsigned u; };
union pk4_u  { bf16_t h[4]; uint2 u; };
union pk8_u  { bf16_t h[8]; uint4 q; };

// A fragment 16(M)x32(K): lane half hf offsets K by 8; VGPRs 4-7 carry K+16.
__device__ __forceinline__ v16bf frag_a(const bf16_t* rowp, int hf) {
  frag_u u;
  u.q[0] = *(const uint4*)(rowp + hf * 8);
  u.q[1] = *(const uint4*)(rowp + 16 + hf * 8);
  return u.v;
}

// B fragment 32(K)x16(N): lane half hf offsets K by 8; VGPRs 4-7 carry K+16.
__device__ __forceinline__ v16bf frag_b(const bf16_t* rowp, int hf) {
  frag_u u;
  u.q[0] = *(const uint4*)(rowp + hf * 8);
  u.q[1] = *(const uint4*)(rowp + 16 + hf * 8);
  return u.v;
}

// One K=64 step: 2 A-frags x (2 B column groups) = 4 WMMA per wave.
#define WMMA_STEP(As_, Bs_)                                          \
  do {                                                               \
    const bf16_t* ar  = (As_) + (16 * rt + ml) * LDA_P;              \
    const bf16_t* br0 = (Bs_) + (32 * ct + ml) * LDB_P;              \
    const bf16_t* br1 = (Bs_) + (32 * ct + 16 + ml) * LDB_P;         \
    v16bf a0 = frag_a(ar, hf);                                       \
    acc0 = wmma_bf16(a0, frag_b(br0, hf), acc0);                     \
    acc1 = wmma_bf16(a0, frag_b(br1, hf), acc1);                     \
    v16bf a1 = frag_a(ar + 32, hf);                                  \
    acc0 = wmma_bf16(a1, frag_b(br0 + 32, hf), acc0);                \
    acc1 = wmma_bf16(a1, frag_b(br1 + 32, hf), acc1);                \
  } while (0)

// Per-lane async global->LDS 16-byte copy (CDNA5, tracked by ASYNCcnt).
__device__ __forceinline__ void async_b128(bf16_t* lds_ptr, const bf16_t* gptr) {
  unsigned loff = (unsigned)(size_t)(__attribute__((address_space(3))) bf16_t*)lds_ptr;
  asm volatile("global_load_async_to_lds_b128 %0, %1, off"
               :: "v"(loff), "v"(gptr) : "memory");
}
__device__ __forceinline__ void wait_async0() {
  asm volatile("s_wait_asynccnt 0" ::: "memory");
}

// ---------------------------------------------------------------------------
// Elementwise fp32 -> bf16 (vectorized: b128 in, b64 out)
// ---------------------------------------------------------------------------
__global__ __launch_bounds__(256) void tobf16_kernel(const float4* __restrict__ in,
                                                     uint2* __restrict__ outp) {
  int i = blockIdx.x * 256 + threadIdx.x;
  float4 v = in[i];
  pk4_u o;
  o.h[0] = (bf16_t)v.x; o.h[1] = (bf16_t)v.y;
  o.h[2] = (bf16_t)v.z; o.h[3] = (bf16_t)v.w;
  outp[i] = o.u;
}

// ---------------------------------------------------------------------------
// Weight transpose: W fp32 [K][N] -> Wt bf16 [N][K]  (once per call, tiny)
// ---------------------------------------------------------------------------
__global__ __launch_bounds__(256) void transpose_w_kernel(const float* __restrict__ W,
                                                          bf16_t* __restrict__ Wt,
                                                          int K, int N) {
  __shared__ bf16_t T[64][65];
  const int k0 = blockIdx.x * 64, n0 = blockIdx.y * 64;
  const int tid = threadIdx.x;
#pragma unroll
  for (int e = 0; e < 16; ++e) {
    int idx = e * 256 + tid; int k = idx >> 6, n = idx & 63;
    T[n][k] = (bf16_t)W[(k0 + k) * N + n0 + n];   // coalesced over n
  }
  __syncthreads();
#pragma unroll
  for (int e = 0; e < 16; ++e) {
    int idx = e * 256 + tid; int n = idx >> 6, k = idx & 63;
    Wt[(n0 + n) * K + k0 + k] = T[n][k];          // coalesced over k
  }
}

// ---------------------------------------------------------------------------
// Projection GEMM: acc = A_f32[MxK] @ Wt^T + bias, scaled by alpha.
// Output: Y bf16 [M][N] (row-major) and/or Yt bf16 [N][ldt] (transposed).
// ---------------------------------------------------------------------------
__global__ __launch_bounds__(256) void gemm_f32A_kernel(
    const float* __restrict__ A, const bf16_t* __restrict__ Wt,
    const float* __restrict__ bias, float alpha,
    bf16_t* __restrict__ Y, bf16_t* __restrict__ Yt, int ldt, int K, int N) {
  __shared__ __align__(16) bf16_t As[64 * LDA_P];
  __shared__ __align__(16) bf16_t Bs[64 * LDB_P];
  const int m0 = blockIdx.x * 64, n0 = blockIdx.y * 64;
  const int tid = threadIdx.x, lane = tid & 31;
  const int rt = (tid >> 5) & 3, ct = tid >> 7;
  const int ml = lane & 15, hf = lane >> 4;
  v8f acc0 = {}, acc1 = {};
  for (int k0 = 0; k0 < K; k0 += 64) {
#pragma unroll
    for (int e = 0; e < 2; ++e) {                       // B: async bf16 row copies
      int c16 = e * 256 + tid;
      int n = c16 >> 3, co = (c16 & 7) * 8;
      async_b128(&Bs[n * LDB_P + co], Wt + (n0 + n) * K + k0 + co);
    }
#pragma unroll
    for (int e = 0; e < 4; ++e) {                       // A: f32x4 -> bf16x4
      int idx = e * 256 + tid; int m = idx >> 4, k4 = (idx & 15) * 4;
      float4 v = *(const float4*)(A + (m0 + m) * K + k0 + k4);
      pk4_u o;
      o.h[0] = (bf16_t)v.x; o.h[1] = (bf16_t)v.y;
      o.h[2] = (bf16_t)v.z; o.h[3] = (bf16_t)v.w;
      *(uint2*)(&As[m * LDA_P + k4]) = o.u;
    }
    wait_async0();
    __syncthreads();
    WMMA_STEP(As, Bs);
    __syncthreads();
  }
  const int c0 = n0 + 32 * ct + ml;
  const float b0v = bias[c0], b1v = bias[c0 + 16];
  const int row0 = m0 + 16 * rt + hf * 8;
  if (Y) {
    int off = row0 * N + c0;
#pragma unroll
    for (int r = 0; r < 8; ++r) {
      Y[off]      = (bf16_t)((acc0[r] + b0v) * alpha);
      Y[off + 16] = (bf16_t)((acc1[r] + b1v) * alpha);
      off += N;
    }
  }
  if (Yt) {                                             // M-contiguous -> uint4
    pk8_u p0, p1;
#pragma unroll
    for (int r = 0; r < 8; ++r) {
      p0.h[r] = (bf16_t)((acc0[r] + b0v) * alpha);
      p1.h[r] = (bf16_t)((acc1[r] + b1v) * alpha);
    }
    *(uint4*)(&Yt[c0 * ldt + row0])        = p0.q;
    *(uint4*)(&Yt[(c0 + 16) * ldt + row0]) = p1.q;
  }
}

// ---------------------------------------------------------------------------
// Generic bf16-A GEMM (A and Wt both async-copied to LDS), concat via
// chunk-aligned split-K pointer select; fused bias/resid epilogue.
// ---------------------------------------------------------------------------
__global__ __launch_bounds__(256) void gemm_bf16A_kernel(
    const bf16_t* __restrict__ A0, int lda0,
    const bf16_t* __restrict__ A1, int lda1, int ksplit,
    const bf16_t* __restrict__ Wt, const float* __restrict__ bias,
    const float* __restrict__ resid,
    float* __restrict__ Yf, bf16_t* __restrict__ Ybf, int K, int N) {
  __shared__ __align__(16) bf16_t As[64 * LDA_P];
  __shared__ __align__(16) bf16_t Bs[64 * LDB_P];
  const int m0 = blockIdx.x * 64, n0 = blockIdx.y * 64;
  const int tid = threadIdx.x, lane = tid & 31;
  const int rt = (tid >> 5) & 3, ct = tid >> 7;
  const int ml = lane & 15, hf = lane >> 4;
  v8f acc0 = {}, acc1 = {};
  for (int k0 = 0; k0 < K; k0 += 64) {
#pragma unroll
    for (int e = 0; e < 2; ++e) {                       // A: async (split-aware)
      int c16 = e * 256 + tid;
      int m = c16 >> 3, co = (c16 & 7) * 8;
      int kg = k0 + co;
      const bf16_t* src = (kg < ksplit) ? A0 + (m0 + m) * lda0 + kg
                                        : A1 + (m0 + m) * lda1 + (kg - ksplit);
      async_b128(&As[m * LDA_P + co], src);
    }
#pragma unroll
    for (int e = 0; e < 2; ++e) {                       // B: async
      int c16 = e * 256 + tid;
      int n = c16 >> 3, co = (c16 & 7) * 8;
      async_b128(&Bs[n * LDB_P + co], Wt + (n0 + n) * K + k0 + co);
    }
    wait_async0();
    __syncthreads();
    WMMA_STEP(As, Bs);
    __syncthreads();
  }
  const int c0 = n0 + 32 * ct + ml;
  const float b0v = bias[c0], b1v = bias[c0 + 16];
  int off = (m0 + 16 * rt + hf * 8) * N + c0;
#pragma unroll
  for (int r = 0; r < 8; ++r) {
    float v0 = acc0[r] + b0v;
    float v1 = acc1[r] + b1v;
    if (resid) { v0 += resid[off]; v1 += resid[off + 16]; }
    if (Yf)  { Yf[off] = v0;           Yf[off + 16] = v1; }
    if (Ybf) { Ybf[off] = (bf16_t)v0;  Ybf[off + 16] = (bf16_t)v1; }
    off += N;
  }
}

// ---------------------------------------------------------------------------
// sim[i,j] = sum_d qk0[i, ecol+d] * qk1[j, ecol+d]   (K = DH = 64, fp32 out)
// Both tiles are raw bf16 row copies -> fully async loads.
// ---------------------------------------------------------------------------
__global__ __launch_bounds__(256) void sim_kernel(
    const bf16_t* __restrict__ q0, const bf16_t* __restrict__ q1,
    float* __restrict__ sim, int ecol, int N1) {
  __shared__ __align__(16) bf16_t As[64 * LDA_P];
  __shared__ __align__(16) bf16_t Bs[64 * LDB_P];
  const int i0 = blockIdx.x * 64, j0 = blockIdx.y * 64;
  const int tid = threadIdx.x, lane = tid & 31;
  const int rt = (tid >> 5) & 3, ct = tid >> 7;
  const int ml = lane & 15, hf = lane >> 4;
#pragma unroll
  for (int e = 0; e < 2; ++e) {
    int c16 = e * 256 + tid;
    int m = c16 >> 3, co = (c16 & 7) * 8;
    async_b128(&As[m * LDA_P + co], q0 + (i0 + m) * 256 + ecol + co);
  }
#pragma unroll
  for (int e = 0; e < 2; ++e) {
    int c16 = e * 256 + tid;
    int j = c16 >> 3, co = (c16 & 7) * 8;
    async_b128(&Bs[j * LDB_P + co], q1 + (j0 + j) * 256 + ecol + co);
  }
  wait_async0();
  __syncthreads();
  v8f acc0 = {}, acc1 = {};
  WMMA_STEP(As, Bs);
  const int c0 = j0 + 32 * ct + ml;
  int off = (i0 + 16 * rt + hf * 8) * N1 + c0;
#pragma unroll
  for (int r = 0; r < 8; ++r) {
    sim[off] = acc0[r];
    sim[off + 16] = acc1[r];
    off += N1;
  }
}

// ---------------------------------------------------------------------------
// Row softmax stats (max & sum of exp), one block per row, float4 sweeps.
// ---------------------------------------------------------------------------
__global__ __launch_bounds__(256) void rowstat_kernel(
    const float* __restrict__ sim, float* __restrict__ rmax,
    float* __restrict__ rsum, int N1) {
  const int i = blockIdx.x, tid = threadIdx.x;
  const float4* row4 = (const float4*)(sim + i * N1);
  const int n4 = N1 >> 2;
  __shared__ float red[256];
  float m = -3.4e38f;
  for (int j = tid; j < n4; j += 256) {
    float4 v = row4[j];
    m = fmaxf(m, fmaxf(fmaxf(v.x, v.y), fmaxf(v.z, v.w)));
  }
  red[tid] = m; __syncthreads();
  for (int s = 128; s > 0; s >>= 1) { if (tid < s) red[tid] = fmaxf(red[tid], red[tid + s]); __syncthreads(); }
  const float mm = red[0]; __syncthreads();
  float s = 0.f;
  for (int j = tid; j < n4; j += 256) {
    float4 v = row4[j];
    s += __expf(v.x - mm) + __expf(v.y - mm) + __expf(v.z - mm) + __expf(v.w - mm);
  }
  red[tid] = s; __syncthreads();
  for (int t = 128; t > 0; t >>= 1) { if (tid < t) red[tid] += red[tid + t]; __syncthreads(); }
  if (tid == 0) { rmax[i] = mm; rsum[i] = red[0]; }
}

// ---------------------------------------------------------------------------
// Column softmax stats, 64 columns per block (coalesced row sweeps).
// ---------------------------------------------------------------------------
__global__ __launch_bounds__(256) void colstat_kernel(
    const float* __restrict__ sim, float* __restrict__ cmax,
    float* __restrict__ csum, int N0, int N1) {
  const int tx = threadIdx.x & 63, ty = threadIdx.x >> 6;
  const int j = blockIdx.x * 64 + tx;
  __shared__ float red[4][64];
  __shared__ float bmax[64];
  float m = -3.4e38f;
  for (int i = ty; i < N0; i += 4) m = fmaxf(m, sim[i * N1 + j]);
  red[ty][tx] = m; __syncthreads();
  if (ty == 0) bmax[tx] = fmaxf(fmaxf(red[0][tx], red[1][tx]), fmaxf(red[2][tx], red[3][tx]));
  __syncthreads();
  const float cm = bmax[tx];
  float s = 0.f;
  for (int i = ty; i < N0; i += 4) s += __expf(sim[i * N1 + j] - cm);
  red[ty][tx] = s; __syncthreads();
  if (ty == 0) { cmax[j] = cm; csum[j] = red[0][tx] + red[1][tx] + red[2][tx] + red[3][tx]; }
}

// ---------------------------------------------------------------------------
// m0[i, ecol+d] = sum_j softmax_row(sim)[i,j] * v1[j, ecol+d]
// A = exp(sim - rowmax) (float4 read, packed uint2 store); B = v1t (async).
// ---------------------------------------------------------------------------
__global__ __launch_bounds__(256) void av_row_kernel(
    const float* __restrict__ sim, const float* __restrict__ rmax,
    const float* __restrict__ rsum, const bf16_t* __restrict__ vt, int ldt,
    bf16_t* __restrict__ mout, int ecol, int N1) {
  __shared__ __align__(16) bf16_t As[64 * LDA_P];
  __shared__ __align__(16) bf16_t Bs[64 * LDB_P];
  const int i0 = blockIdx.x * 64;
  const int tid = threadIdx.x, lane = tid & 31;
  const int rt = (tid >> 5) & 3, ct = tid >> 7;
  const int ml = lane & 15, hf = lane >> 4;
  v8f acc0 = {}, acc1 = {};
  for (int j0 = 0; j0 < N1; j0 += 64) {
#pragma unroll
    for (int e = 0; e < 2; ++e) {                       // B: v1t rows, async
      int c16 = e * 256 + tid;
      int n = c16 >> 3, co = (c16 & 7) * 8;
      async_b128(&Bs[n * LDB_P + co], vt + (ecol + n) * ldt + j0 + co);
    }
#pragma unroll
    for (int e = 0; e < 4; ++e) {                       // A: exp(sim - max)
      int idx = e * 256 + tid; int m = idx >> 4, k4 = (idx & 15) * 4;
      float4 sv = *(const float4*)(sim + (i0 + m) * N1 + j0 + k4);
      float rm = rmax[i0 + m];
      pk4_u o;
      o.h[0] = (bf16_t)__expf(sv.x - rm); o.h[1] = (bf16_t)__expf(sv.y - rm);
      o.h[2] = (bf16_t)__expf(sv.z - rm); o.h[3] = (bf16_t)__expf(sv.w - rm);
      *(uint2*)(&As[m * LDA_P + k4]) = o.u;
    }
    wait_async0();
    __syncthreads();
    WMMA_STEP(As, Bs);
    __syncthreads();
  }
  const int c0 = 32 * ct + ml;
  const int row0 = i0 + 16 * rt + hf * 8;
  int off = row0 * 256 + ecol + c0;
#pragma unroll
  for (int r = 0; r < 8; ++r) {
    float inv = 1.0f / rsum[row0 + r];
    mout[off]      = (bf16_t)(acc0[r] * inv);
    mout[off + 16] = (bf16_t)(acc1[r] * inv);
    off += 256;
  }
}

// ---------------------------------------------------------------------------
// m1[j, ecol+d] = sum_i softmax_col(sim)[i,j] * v0[i, ecol+d]
// A = P^T (pair-packed transposed LDS writes); B = v0t (async).
// ---------------------------------------------------------------------------
__global__ __launch_bounds__(256) void av_col_kernel(
    const float* __restrict__ sim, const float* __restrict__ cmax,
    const float* __restrict__ csum, const bf16_t* __restrict__ vt, int ldt,
    bf16_t* __restrict__ mout, int ecol, int N0, int N1) {
  __shared__ __align__(16) bf16_t As[64 * LDA_P];
  __shared__ __align__(16) bf16_t Bs[64 * LDB_P];
  const int j0 = blockIdx.x * 64;
  const int tid = threadIdx.x, lane = tid & 31;
  const int rt = (tid >> 5) & 3, ct = tid >> 7;
  const int ml = lane & 15, hf = lane >> 4;
  v8f acc0 = {}, acc1 = {};
  for (int i0 = 0; i0 < N0; i0 += 64) {
#pragma unroll
    for (int e = 0; e < 2; ++e) {                       // B: v0t rows, async
      int c16 = e * 256 + tid;
      int n = c16 >> 3, co = (c16 & 7) * 8;
      async_b128(&Bs[n * LDB_P + co], vt + (ecol + n) * ldt + i0 + co);
    }
#pragma unroll
    for (int e = 0; e < 8; ++e) {                       // A: P^T, 2 k per store
      int idx = e * 256 + tid; int ik = (idx >> 6) * 2, jj = idx & 63;
      float cm = cmax[j0 + jj];
      float s0 = sim[(i0 + ik) * N1 + j0 + jj];
      float s1 = sim[(i0 + ik + 1) * N1 + j0 + jj];
      pk2_u o;
      o.h[0] = (bf16_t)__expf(s0 - cm);
      o.h[1] = (bf16_t)__expf(s1 - cm);
      *(unsigned*)(&As[jj * LDA_P + ik]) = o.u;
    }
    wait_async0();
    __syncthreads();
    WMMA_STEP(As, Bs);
    __syncthreads();
  }
  const int c0 = 32 * ct + ml;
  const int row0 = j0 + 16 * rt + hf * 8;
  int off = row0 * 256 + ecol + c0;
#pragma unroll
  for (int r = 0; r < 8; ++r) {
    float inv = 1.0f / csum[row0 + r];
    mout[off]      = (bf16_t)(acc0[r] * inv);
    mout[off + 16] = (bf16_t)(acc1[r] * inv);
    off += 256;
  }
}

// ---------------------------------------------------------------------------
// LayerNorm(512) + exact GELU, one block (256 threads) per row -> bf16.
// ---------------------------------------------------------------------------
__global__ __launch_bounds__(256) void ln_gelu_kernel(
    const float* __restrict__ hin, const float* __restrict__ g,
    const float* __restrict__ bta, bf16_t* __restrict__ hout) {
  const int row = blockIdx.x, tid = threadIdx.x;
  const float* x = hin + row * 512;
  float a0 = x[tid], a1 = x[tid + 256];
  __shared__ float red[256];
  red[tid] = a0 + a1; __syncthreads();
  for (int t = 128; t > 0; t >>= 1) { if (tid < t) red[tid] += red[tid + t]; __syncthreads(); }
  const float mu = red[0] * (1.0f / 512.0f); __syncthreads();
  float d0 = a0 - mu, d1 = a1 - mu;
  red[tid] = d0 * d0 + d1 * d1; __syncthreads();
  for (int t = 128; t > 0; t >>= 1) { if (tid < t) red[tid] += red[tid + t]; __syncthreads(); }
  const float rstd = rsqrtf(red[0] * (1.0f / 512.0f) + 1e-5f);
  float y0 = d0 * rstd * g[tid] + bta[tid];
  y0 = 0.5f * y0 * (1.0f + erff(y0 * 0.70710678118f));
  float y1 = d1 * rstd * g[tid + 256] + bta[tid + 256];
  y1 = 0.5f * y1 * (1.0f + erff(y1 * 0.70710678118f));
  hout[row * 512 + tid]       = (bf16_t)y0;
  hout[row * 512 + tid + 256] = (bf16_t)y1;
}

// ---------------------------------------------------------------------------
// Host orchestration
// ---------------------------------------------------------------------------
extern "C" void kernel_launch(void* const* d_in, const int* in_sizes, int n_in,
                              void* d_out, int out_size, void* d_ws, size_t ws_size,
                              hipStream_t stream) {
  (void)in_sizes; (void)n_in; (void)out_size; (void)ws_size;
  const float* x0     = (const float*)d_in[0];
  const float* x1     = (const float*)d_in[1];
  const float* qk_w   = (const float*)d_in[2];
  const float* qk_b   = (const float*)d_in[3];
  const float* v_w    = (const float*)d_in[4];
  const float* v_b    = (const float*)d_in[5];
  const float* out_w  = (const float*)d_in[6];
  const float* out_b  = (const float*)d_in[7];
  const float* ffn_w1 = (const float*)d_in[8];
  const float* ffn_b1 = (const float*)d_in[9];
  const float* ln_g   = (const float*)d_in[10];
  const float* ln_b   = (const float*)d_in[11];
  const float* ffn_w2 = (const float*)d_in[12];
  const float* ffn_b2 = (const float*)d_in[13];
  float* out = (float*)d_out;

  constexpr int B = 2, N = 2048, E = 256, H = 4, DH = 64;
  const int M = B * N;                       // 4096 rows
  const float SCALE = 0.3535533905932738f;   // 64^-0.25

  char* ws = (char*)d_ws;
  size_t off = 0;
  auto alloc = [&](size_t bytes) -> void* {
    void* p = ws + off; off = (off + bytes + 255) & ~(size_t)255; return p;
  };
  bf16_t* qk0b = (bf16_t*)alloc((size_t)M * E * 2);
  bf16_t* qk1b = (bf16_t*)alloc((size_t)M * E * 2);
  bf16_t* v0t  = (bf16_t*)alloc((size_t)E * M * 2);   // [E][M] transposed V
  bf16_t* v1t  = (bf16_t*)alloc((size_t)E * M * 2);
  bf16_t* xb0  = (bf16_t*)alloc((size_t)M * E * 2);
  bf16_t* xb1  = (bf16_t*)alloc((size_t)M * E * 2);
  bf16_t* m0b  = (bf16_t*)alloc((size_t)M * E * 2);
  bf16_t* m1b  = (bf16_t*)alloc((size_t)M * E * 2);
  bf16_t* mp0b = (bf16_t*)alloc((size_t)M * E * 2);
  bf16_t* mp1b = (bf16_t*)alloc((size_t)M * E * 2);
  bf16_t* qkwt = (bf16_t*)alloc((size_t)E * E * 2);   // bf16 [N][K] weights
  bf16_t* vwt  = (bf16_t*)alloc((size_t)E * E * 2);
  bf16_t* owt  = (bf16_t*)alloc((size_t)E * E * 2);
  bf16_t* w1t  = (bf16_t*)alloc((size_t)2 * E * 2 * E * 2);
  bf16_t* w2t  = (bf16_t*)alloc((size_t)E * 2 * E * 2);
  float*  simb = (float*)alloc((size_t)N * N * 4);    // one (b,h) slice, L2-hot
  float*  rmax = (float*)alloc((size_t)N * 4);
  float*  rsum = (float*)alloc((size_t)N * 4);
  float*  cmax = (float*)alloc((size_t)N * 4);
  float*  csum = (float*)alloc((size_t)N * 4);
  float*  hbuf = (float*)alloc((size_t)M * 2 * E * 4);
  bf16_t* hb   = (bf16_t*)alloc((size_t)M * 2 * E * 2);

  dim3 blk(256);

  // one-time per call: bf16 x copies + bf16 transposed weights
  tobf16_kernel<<<(M * E) / 1024, blk, 0, stream>>>((const float4*)x0, (uint2*)xb0);
  tobf16_kernel<<<(M * E) / 1024, blk, 0, stream>>>((const float4*)x1, (uint2*)xb1);
  transpose_w_kernel<<<dim3(E / 64, E / 64), blk, 0, stream>>>(qk_w, qkwt, E, E);
  transpose_w_kernel<<<dim3(E / 64, E / 64), blk, 0, stream>>>(v_w, vwt, E, E);
  transpose_w_kernel<<<dim3(E / 64, E / 64), blk, 0, stream>>>(out_w, owt, E, E);
  transpose_w_kernel<<<dim3(2 * E / 64, 2 * E / 64), blk, 0, stream>>>(ffn_w1, w1t, 2 * E, 2 * E);
  transpose_w_kernel<<<dim3(2 * E / 64, E / 64), blk, 0, stream>>>(ffn_w2, w2t, 2 * E, E);

  // projections: qk (scaled, row-major) and v (stored transposed [E][M])
  dim3 gp(M / 64, E / 64);
  gemm_f32A_kernel<<<gp, blk, 0, stream>>>(x0, qkwt, qk_b, SCALE, qk0b, nullptr, 0, E, E);
  gemm_f32A_kernel<<<gp, blk, 0, stream>>>(x1, qkwt, qk_b, SCALE, qk1b, nullptr, 0, E, E);
  gemm_f32A_kernel<<<gp, blk, 0, stream>>>(x0, vwt, v_b, 1.0f, nullptr, v0t, M, E, E);
  gemm_f32A_kernel<<<gp, blk, 0, stream>>>(x1, vwt, v_b, 1.0f, nullptr, v1t, M, E, E);

  // attention, one (b,h) slice at a time (sim slice stays hot in 192MB L2)
  for (int b = 0; b < B; ++b) {
    const bf16_t* q0s = qk0b + (size_t)b * N * E;
    const bf16_t* q1s = qk1b + (size_t)b * N * E;
    const bf16_t* v0s = v0t + (size_t)b * N;   // [E][M] slice: col offset b*N
    const bf16_t* v1s = v1t + (size_t)b * N;
    bf16_t* m0s = m0b + (size_t)b * N * E;
    bf16_t* m1s = m1b + (size_t)b * N * E;
    for (int h = 0; h < H; ++h) {
      const int ecol = h * DH;
      dim3 gs(N / 64, N / 64);
      sim_kernel<<<gs, blk, 0, stream>>>(q0s, q1s, simb, ecol, N);
      rowstat_kernel<<<N, blk, 0, stream>>>(simb, rmax, rsum, N);
      colstat_kernel<<<N / 64, blk, 0, stream>>>(simb, cmax, csum, N, N);
      av_row_kernel<<<N / 64, blk, 0, stream>>>(simb, rmax, rsum, v1s, M, m0s, ecol, N);
      av_col_kernel<<<N / 64, blk, 0, stream>>>(simb, cmax, csum, v0s, M, m1s, ecol, N, N);
    }
  }

  // out projection -> bf16
  gemm_bf16A_kernel<<<gp, blk, 0, stream>>>(m0b, E, m0b, E, E, owt, out_b,
                                            nullptr, nullptr, mp0b, E, E);
  gemm_bf16A_kernel<<<gp, blk, 0, stream>>>(m1b, E, m1b, E, E, owt, out_b,
                                            nullptr, nullptr, mp1b, E, E);

  // FFN for input 0
  dim3 gf1(M / 64, (2 * E) / 64);
  dim3 gf2(M / 64, E / 64);
  gemm_bf16A_kernel<<<gf1, blk, 0, stream>>>(xb0, E, mp0b, E, E, w1t, ffn_b1,
                                             nullptr, hbuf, nullptr, 2 * E, 2 * E);
  ln_gelu_kernel<<<M, blk, 0, stream>>>(hbuf, ln_g, ln_b, hb);
  gemm_bf16A_kernel<<<gf2, blk, 0, stream>>>(hb, 2 * E, hb, 2 * E, 2 * E, w2t, ffn_b2,
                                             x0, out, nullptr, 2 * E, E);

  // FFN for input 1
  gemm_bf16A_kernel<<<gf1, blk, 0, stream>>>(xb1, E, mp1b, E, E, w1t, ffn_b1,
                                             nullptr, hbuf, nullptr, 2 * E, 2 * E);
  ln_gelu_kernel<<<M, blk, 0, stream>>>(hbuf, ln_g, ln_b, hb);
  gemm_bf16A_kernel<<<gf2, blk, 0, stream>>>(hb, 2 * E, hb, 2 * E, 2 * E, w2t, ffn_b2,
                                             x1, out + M * E, nullptr, 2 * E, E);
}
